// MultiHeadAttention_40767829573859
// MI455X (gfx1250) — compile-verified, matrix operand order checked by reading, not run on hardware
//
#include <hip/hip_runtime.h>
#include <hip/hip_bf16.h>
#include <math.h>

#define BATCH  4
#define SEQ    2048
#define DMODEL 512
#define NHEAD  8
#define DK     64
#define LNEPS  1e-5f

typedef __attribute__((ext_vector_type(16))) _Float16 v16h;
typedef __attribute__((ext_vector_type(8)))  float    v8f;

// ---------------------------------------------------------------------------
// CDNA5 async global->LDS copy (ASYNCcnt-tracked, bypasses VGPRs).
// Each lane moves 16 contiguous bytes: LDS[vdst] = MEM[vaddr].
// Generic pointer low 32 bits == wave-relative LDS offset (ISA 10.2).
// ---------------------------------------------------------------------------
__device__ __forceinline__ void async_load_b128(const void* gaddr, void* lptr) {
  unsigned lds_off = (unsigned)(unsigned long long)lptr;
  asm volatile("global_load_async_to_lds_b128 %0, %1, off"
               :: "v"(lds_off), "v"((unsigned long long)gaddr) : "memory");
}
__device__ __forceinline__ void wait_async0() {
  asm volatile("s_wait_asynccnt 0x0" ::: "memory");
}

// ---------------------------------------------------------------------------
// Fragment loader for V_WMMA_F32_16X16X32_F16 from a 16-row LDS tile,
// row-major, leading dim lda (halfs).  16-bit A layout (ISA 7.12.2):
// lane L: row=L&15, half=L>>4; e<8 -> K=half*8+e ; e>=8 -> K=16+half*8+(e-8).
// B is fed from a transposed LDS tile (Bt[col][K]) with the same loader.
// ---------------------------------------------------------------------------
__device__ __forceinline__ v16h load_frag(const _Float16* __restrict__ base,
                                          int lda, int k0) {
  const int lane = threadIdx.x & 31;
  const int r    = lane & 15;
  const int half = lane >> 4;
  const _Float16* p = base + r * lda + k0 + half * 8;
  v16h f;
#pragma unroll
  for (int e = 0; e < 8; ++e) f[e] = p[e];
#pragma unroll
  for (int e = 0; e < 8; ++e) f[8 + e] = p[16 + e];
  return f;
}

#define WMMA16(acc, a, b) \
  acc = __builtin_amdgcn_wmma_f32_16x16x32_f16(false, a, false, b, (short)0, acc, false, false)

// ---------------------------------------------------------------------------
__global__ void cast_f16(const float* __restrict__ src, _Float16* __restrict__ dst,
                         int n, float scale) {
  int i = blockIdx.x * blockDim.x + threadIdx.x;
  if (i < n) dst[i] = (_Float16)(src[i] * scale);
}

// ---------------------------------------------------------------------------
// C[M,512] = A[M,512](f16) x W[512,512](f16) + biasScale*bias.
// Block 256 thr = 8 waves (4x2); wave tile 32x32 (2x2 WMMA); block 128x64.
// A tile staged by async global->LDS b128; W tile transposed by ds stores.
// Double-buffered: prefetch k0+32 while computing k0.
// ---------------------------------------------------------------------------
template <typename OutT>
__global__ void gemm512(const _Float16* __restrict__ A,
                        const _Float16* __restrict__ W,
                        const float* __restrict__ bias, float biasScale,
                        OutT* __restrict__ C) {
  __shared__ _Float16 lA[2][128][32 + 8];
  __shared__ _Float16 lBt[2][64][32 + 8];
  const int tid  = threadIdx.x;
  const int wave = tid >> 5;
  const int rowBase0 = blockIdx.y * 128;
  const int colBase0 = blockIdx.x * 64;
  const int wRow = (wave >> 1) * 32;
  const int wCol = (wave & 1) * 32;

  auto stage = [&](int buf, int k0) {
#pragma unroll
    for (int i = 0; i < 2; ++i) {                    // A: 128x32 halfs, 512 b128 chunks
      int chunk = tid + i * 256;
      int r = chunk >> 2, kk = (chunk & 3) << 3;
      async_load_b128(A + (size_t)(rowBase0 + r) * DMODEL + k0 + kk, &lA[buf][r][kk]);
    }
#pragma unroll
    for (int i = 0; i < 8; ++i) {                    // W: 32x64 -> transposed [col][k]
      int idx = tid + i * 256;
      int c = idx & 63, kk = idx >> 6;
      lBt[buf][c][kk] = W[(size_t)(k0 + kk) * DMODEL + colBase0 + c];
    }
  };

  v8f a00 = {}, a01 = {}, a10 = {}, a11 = {};
  stage(0, 0);
  int buf = 0;
  for (int k0 = 0; k0 < DMODEL; k0 += 32) {
    wait_async0();
    __syncthreads();
    if (k0 + 32 < DMODEL) stage(buf ^ 1, k0 + 32);
    v16h af0 = load_frag(&lA[buf][wRow][0],      32 + 8, 0);
    v16h af1 = load_frag(&lA[buf][wRow + 16][0], 32 + 8, 0);
    v16h bf0 = load_frag(&lBt[buf][wCol][0],      32 + 8, 0);
    v16h bf1 = load_frag(&lBt[buf][wCol + 16][0], 32 + 8, 0);
    WMMA16(a00, af0, bf0); WMMA16(a01, af0, bf1);
    WMMA16(a10, af1, bf0); WMMA16(a11, af1, bf1);
    buf ^= 1;
  }
  const int lane = tid & 31;
  const int cl   = lane & 15;
  const int rOff = (lane >> 4) * 8;
  v8f accs[4] = {a00, a01, a10, a11};
#pragma unroll
  for (int t = 0; t < 4; ++t) {
    int row0 = rowBase0 + wRow + (t >> 1) * 16 + rOff;
    int col  = colBase0 + wCol + (t & 1) * 16 + cl;
    float bv = bias[col] * biasScale;
#pragma unroll
    for (int e = 0; e < 8; ++e)
      C[(size_t)(row0 + e) * DMODEL + col] = (OutT)(accs[t][e] + bv);
  }
}

// ---------------------------------------------------------------------------
// Masked scores: S[b,h,i,j] = sum_d q[b,i,h,d]*k[b,j,h,d]  (1/sqrt(dk) folded
// into q).  Block tile 128(rows) x 64(cols), K=64: both q and k tiles are
// row-contiguous -> fully async-staged; 8 WMMAs per wave.
// ---------------------------------------------------------------------------
__global__ void scores_kernel(const _Float16* __restrict__ q,
                              const _Float16* __restrict__ k,
                              const unsigned char* __restrict__ mask,
                              float* __restrict__ attn) {
  __shared__ _Float16 lA[128][64 + 8];   // q rows [row][d]
  __shared__ _Float16 lBt[64][64 + 8];   // k rows [col][d] (== B transposed)
  const int tid  = threadIdx.x;
  const int wave = tid >> 5;
  const int b = blockIdx.z >> 3, h = blockIdx.z & 7;
  const int rowBase0 = blockIdx.y * 128;
  const int colBase0 = blockIdx.x * 64;
  const int wRow = (wave >> 1) * 32;
  const int wCol = (wave & 1) * 32;
  const size_t headBase = ((size_t)b * SEQ) * DMODEL + (size_t)h * DK;
#pragma unroll
  for (int i = 0; i < 4; ++i) {                      // q: 128x64 halfs, 1024 chunks
    int chunk = tid + i * 256;
    int r = chunk >> 3, dd = (chunk & 7) << 3;
    async_load_b128(q + headBase + (size_t)(rowBase0 + r) * DMODEL + dd, &lA[r][dd]);
  }
#pragma unroll
  for (int i = 0; i < 2; ++i) {                      // k: 64x64 halfs, 512 chunks
    int chunk = tid + i * 256;
    int c = chunk >> 3, dd = (chunk & 7) << 3;
    async_load_b128(k + headBase + (size_t)(colBase0 + c) * DMODEL + dd, &lBt[c][dd]);
  }
  wait_async0();
  __syncthreads();
  v8f a00 = {}, a01 = {}, a10 = {}, a11 = {};
#pragma unroll
  for (int k0 = 0; k0 < DK; k0 += 32) {
    v16h af0 = load_frag(&lA[wRow][0],      64 + 8, k0);
    v16h af1 = load_frag(&lA[wRow + 16][0], 64 + 8, k0);
    v16h bf0 = load_frag(&lBt[wCol][0],      64 + 8, k0);
    v16h bf1 = load_frag(&lBt[wCol + 16][0], 64 + 8, k0);
    WMMA16(a00, af0, bf0); WMMA16(a01, af0, bf1);
    WMMA16(a10, af1, bf0); WMMA16(a11, af1, bf1);
  }
  const int lane = tid & 31;
  const int cl   = lane & 15;
  const int rOff = (lane >> 4) * 8;
  const size_t mBase = (size_t)b * SEQ * SEQ;
  const size_t aBase = (size_t)(b * NHEAD + h) * SEQ * SEQ;
  v8f accs[4] = {a00, a01, a10, a11};
#pragma unroll
  for (int t = 0; t < 4; ++t) {
    int row0 = rowBase0 + wRow + (t >> 1) * 16 + rOff;
    int col  = colBase0 + wCol + (t & 1) * 16 + cl;
#pragma unroll
    for (int e = 0; e < 8; ++e) {
      int row = row0 + e;
      float s = accs[t][e];
      if (mask[mBase + (size_t)row * SEQ + col]) s = -1e9f;
      attn[aBase + (size_t)row * SEQ + col] = s;
    }
  }
}

// ---------------------------------------------------------------------------
// In-place row softmax over 2048-wide rows; one 256-thread block per row.
// ---------------------------------------------------------------------------
__global__ void softmax_rows(float* __restrict__ attn) {
  __shared__ float red[256];
  float* p = attn + (size_t)blockIdx.x * SEQ;
  const int tid = threadIdx.x;
  float vals[8];
  float m = -3.402823e38f;
#pragma unroll
  for (int i = 0; i < 8; ++i) { vals[i] = p[tid + i * 256]; m = fmaxf(m, vals[i]); }
  red[tid] = m; __syncthreads();
  for (int s = 128; s > 0; s >>= 1) {
    if (tid < s) red[tid] = fmaxf(red[tid], red[tid + s]);
    __syncthreads();
  }
  m = red[0]; __syncthreads();
  float sum = 0.f;
#pragma unroll
  for (int i = 0; i < 8; ++i) { vals[i] = __expf(vals[i] - m); sum += vals[i]; }
  red[tid] = sum; __syncthreads();
  for (int s = 128; s > 0; s >>= 1) {
    if (tid < s) red[tid] += red[tid + s];
    __syncthreads();
  }
  const float inv = 1.f / red[0];
#pragma unroll
  for (int i = 0; i < 8; ++i) p[tid + i * 256] = vals[i] * inv;
}

// ---------------------------------------------------------------------------
// Context: ctx[b,i,h*64+d] = sum_j attn[b,h,i,j]*v[b,j,h*64+d]
// M=2048, N=64, K=2048 per (b,h).  Block tile 128x64, wave tile 32x32.
// attn cast f32->f16 during LDS staging (so no async path here).
// ---------------------------------------------------------------------------
__global__ void context_kernel(const float* __restrict__ attn,
                               const _Float16* __restrict__ v,
                               _Float16* __restrict__ ctx) {
  __shared__ _Float16 lA[128][32 + 8];
  __shared__ _Float16 lBt[64][32 + 8];
  const int tid  = threadIdx.x;
  const int wave = tid >> 5;
  const int b = blockIdx.z >> 3, h = blockIdx.z & 7;
  const int rowBase0 = blockIdx.y * 128;
  const int wRow = (wave >> 1) * 32;
  const int wCol = (wave & 1) * 32;
  const size_t aBase = (size_t)(b * NHEAD + h) * SEQ * SEQ;
  const size_t vBase = ((size_t)b * SEQ) * DMODEL + (size_t)h * DK;
  v8f a00 = {}, a01 = {}, a10 = {}, a11 = {};
  for (int k0 = 0; k0 < SEQ; k0 += 32) {
    __syncthreads();
#pragma unroll
    for (int i = 0; i < 16; ++i) {                   // attn: 128x32, f32->f16
      int idx = tid + i * 256;
      int kk = idx & 31, r = idx >> 5;
      lA[r][kk] = (_Float16)attn[aBase + (size_t)(rowBase0 + r) * SEQ + k0 + kk];
    }
#pragma unroll
    for (int i = 0; i < 8; ++i) {                    // v: 32x64 -> transposed
      int idx = tid + i * 256;
      int c = idx & 63, kk = idx >> 6;
      lBt[c][kk] = v[vBase + (size_t)(k0 + kk) * DMODEL + c];
    }
    __syncthreads();
    v16h af0 = load_frag(&lA[wRow][0],      32 + 8, 0);
    v16h af1 = load_frag(&lA[wRow + 16][0], 32 + 8, 0);
    v16h bf0 = load_frag(&lBt[wCol][0],      32 + 8, 0);
    v16h bf1 = load_frag(&lBt[wCol + 16][0], 32 + 8, 0);
    WMMA16(a00, af0, bf0); WMMA16(a01, af0, bf1);
    WMMA16(a10, af1, bf0); WMMA16(a11, af1, bf1);
  }
  const int lane = tid & 31;
  const int cl   = lane & 15;
  const int rOff = (lane >> 4) * 8;
  v8f accs[4] = {a00, a01, a10, a11};
#pragma unroll
  for (int t = 0; t < 4; ++t) {
    int row0 = rowBase0 + wRow + (t >> 1) * 16 + rOff;
    int col  = wCol + (t & 1) * 16 + cl;             // 0..63 within d_k
#pragma unroll
    for (int e = 0; e < 8; ++e)
      ctx[((size_t)b * SEQ + row0 + e) * DMODEL + h * DK + col] = (_Float16)accs[t][e];
  }
}

// ---------------------------------------------------------------------------
// out = LayerNorm(proj + residual), rows of 512; one block per row.
// ---------------------------------------------------------------------------
__global__ void add_ln(const float* __restrict__ proj,
                       const float* __restrict__ resid,
                       float* __restrict__ out) {
  __shared__ float red[256];
  const size_t row = blockIdx.x;
  const float* p = proj  + row * DMODEL;
  const float* r = resid + row * DMODEL;
  float* o = out + row * DMODEL;
  const int tid = threadIdx.x;
  float x0 = p[tid] + r[tid];
  float x1 = p[tid + 256] + r[tid + 256];
  red[tid] = x0 + x1; __syncthreads();
  for (int s = 128; s > 0; s >>= 1) {
    if (tid < s) red[tid] += red[tid + s];
    __syncthreads();
  }
  const float mu = red[0] * (1.f / DMODEL); __syncthreads();
  float d0 = x0 - mu, d1 = x1 - mu;
  red[tid] = d0 * d0 + d1 * d1; __syncthreads();
  for (int s = 128; s > 0; s >>= 1) {
    if (tid < s) red[tid] += red[tid + s];
    __syncthreads();
  }
  const float inv = rsqrtf(red[0] * (1.f / DMODEL) + LNEPS);
  o[tid]       = d0 * inv;
  o[tid + 256] = d1 * inv;
}

// ---------------------------------------------------------------------------
extern "C" void kernel_launch(void* const* d_in, const int* in_sizes, int n_in,
                              void* d_out, int out_size, void* d_ws, size_t ws_size,
                              hipStream_t stream) {
  (void)in_sizes; (void)n_in; (void)out_size; (void)ws_size;
  const float* Q  = (const float*)d_in[0];
  const float* K  = (const float*)d_in[1];
  const float* V  = (const float*)d_in[2];
  const unsigned char* mask = (const unsigned char*)d_in[3];
  const float* Wq = (const float*)d_in[4];
  const float* bq = (const float*)d_in[5];
  const float* Wk = (const float*)d_in[6];
  const float* bk = (const float*)d_in[7];
  const float* Wv = (const float*)d_in[8];
  const float* bv = (const float*)d_in[9];
  const float* Wo = (const float*)d_in[10];
  const float* bo = (const float*)d_in[11];

  const int    M    = BATCH * SEQ;                 // 8192
  const size_t nQKV = (size_t)M * DMODEL;          // 4,194,304
  const size_t nW   = (size_t)DMODEL * DMODEL;     // 262,144

  char* w = (char*)d_ws;
  _Float16* hQ   = (_Float16*)w; w += nQKV * 2;
  _Float16* hK   = (_Float16*)w; w += nQKV * 2;
  _Float16* hV   = (_Float16*)w; w += nQKV * 2;
  _Float16* hWq  = (_Float16*)w; w += nW * 2;
  _Float16* hWk  = (_Float16*)w; w += nW * 2;
  _Float16* hWv  = (_Float16*)w; w += nW * 2;
  _Float16* hWo  = (_Float16*)w; w += nW * 2;
  _Float16* hq   = (_Float16*)w; w += nQKV * 2;
  _Float16* hk   = (_Float16*)w; w += nQKV * 2;
  _Float16* hv   = (_Float16*)w; w += nQKV * 2;
  _Float16* hctx = (_Float16*)w; w += nQKV * 2;
  float*    proj = (float*)w;    w += nQKV * 4;    // ~74 MB total

  float* out  = (float*)d_out;
  float* attn = out + nQKV;                        // [B,H,S,S] f32

  const dim3 blk(256);
  const float qs = 0.125f;                         // 1/sqrt(d_k), folded into Wq,bq

  cast_f16<<<(int)((nQKV + 255) / 256), blk, 0, stream>>>(Q,  hQ,  (int)nQKV, 1.f);
  cast_f16<<<(int)((nQKV + 255) / 256), blk, 0, stream>>>(K,  hK,  (int)nQKV, 1.f);
  cast_f16<<<(int)((nQKV + 255) / 256), blk, 0, stream>>>(V,  hV,  (int)nQKV, 1.f);
  cast_f16<<<(int)((nW   + 255) / 256), blk, 0, stream>>>(Wq, hWq, (int)nW,  qs);
  cast_f16<<<(int)((nW   + 255) / 256), blk, 0, stream>>>(Wk, hWk, (int)nW,  1.f);
  cast_f16<<<(int)((nW   + 255) / 256), blk, 0, stream>>>(Wv, hWv, (int)nW,  1.f);
  cast_f16<<<(int)((nW   + 255) / 256), blk, 0, stream>>>(Wo, hWo, (int)nW,  1.f);

  const dim3 gProj(DMODEL / 64, M / 128);                 // (8, 64)
  gemm512<_Float16><<<gProj, blk, 0, stream>>>(hQ, hWq, bq, qs,  hq);
  gemm512<_Float16><<<gProj, blk, 0, stream>>>(hK, hWk, bk, 1.f, hk);
  gemm512<_Float16><<<gProj, blk, 0, stream>>>(hV, hWv, bv, 1.f, hv);

  const dim3 gScore(SEQ / 64, SEQ / 128, BATCH * NHEAD);  // (32, 16, 32)
  scores_kernel<<<gScore, blk, 0, stream>>>(hq, hk, mask, attn);

  softmax_rows<<<BATCH * NHEAD * SEQ, blk, 0, stream>>>(attn);

  const dim3 gCtx(1, SEQ / 128, BATCH * NHEAD);           // (1, 16, 32)
  context_kernel<<<gCtx, blk, 0, stream>>>(attn, hv, hctx);

  gemm512<float><<<gProj, blk, 0, stream>>>(hctx, hWo, bo, 1.f, proj);

  add_ln<<<M, blk, 0, stream>>>(proj, Q, out);
}
